// Qwen3_5Attention_2070174236857
// MI455X (gfx1250) — compile-verified, matrix operand order checked by reading, not run on hardware
//
#include <hip/hip_runtime.h>
#include <hip/hip_bf16.h>
#include <math.h>

// ---------------------------------------------------------------------------
// Qwen3.5-style attention layer for MI455X (gfx1250), bf16 WMMA everywhere.
// Round 3: double-buffered GEMM operands (kills WAR v_nops + full loadcnt
// stalls seen in round-2 asm); prefetch next K/V block in attention.
// ---------------------------------------------------------------------------

typedef __attribute__((ext_vector_type(16))) __bf16 bf16x16;
typedef __attribute__((ext_vector_type(8)))  float  f32x8;

struct U8 { uint4 lo, hi; };

static constexpr int T_   = 2048;
static constexpr int HID_ = 2048;
static constexpr int H_   = 16;
static constexpr int KV_  = 8;
static constexpr int D_   = 128;
static constexpr int HALF_ = D_ / 2;          // 64
static constexpr int QGN_  = H_ * 2 * D_;     // 4096 (q|gate per head)
static constexpr int KVN_  = KV_ * D_;        // 1024
static constexpr int OD_   = H_ * D_;         // 2048

__device__ __forceinline__ __bf16 tobf(float f) {
  unsigned u = __builtin_bit_cast(unsigned, f);
  u += 0x7FFFu + ((u >> 16) & 1u);            // round-to-nearest-even
  unsigned short h = (unsigned short)(u >> 16);
  return __builtin_bit_cast(__bf16, h);
}

// Load 16 bf16 in WMMA operand order: halves [0..7] from p, [8..15] from p+16.
__device__ __forceinline__ bf16x16 load16(const __bf16* p) {
  U8 u;
  u.lo = *(const uint4*)(p);
  u.hi = *(const uint4*)(p + 16);
  return __builtin_bit_cast(bf16x16, u);
}

__device__ __forceinline__ f32x8 wmma_bf16(bf16x16 a, bf16x16 b, f32x8 c) {
  return __builtin_amdgcn_wmma_f32_16x16x32_bf16(false, a, false, b,
                                                 (short)0, c, false, false);
}

// ---------------------------------------------------------------------------
// fp32 -> bf16 flat convert
__global__ void cvt_bf16(const float* __restrict__ in, __bf16* __restrict__ out,
                         size_t n) {
  size_t i = (size_t)blockIdx.x * blockDim.x + threadIdx.x;
  if (i < n) out[i] = tobf(in[i]);
}

// W (K x N fp32, row-major) -> WT (N x K bf16, row-major): out[n*K+k] = in[k*N+n]
__global__ void transpose_cvt(const float* __restrict__ in, __bf16* __restrict__ out,
                              int K, int N) {
  size_t i = (size_t)blockIdx.x * blockDim.x + threadIdx.x;
  if (i >= (size_t)K * N) return;
  size_t n = i / (size_t)K;
  size_t k = i % (size_t)K;
  out[i] = tobf(in[k * (size_t)N + n]);
}

// ---------------------------------------------------------------------------
// Generic C(MxN,f32) = A(MxK,bf16 row-major) * BT(NxK,bf16 row-major)^T
// One wave computes a 32x64 tile: 2 A-tiles x 4 B-tiles, 8 accumulators.
// Operands double-buffered: next k-step's 6 load16 issued before this step's
// 8 WMMAs, so loads overlap WMMA latency and no WAR hazard NOPs are needed.
__global__ void gemm_bf16(const __bf16* __restrict__ A, const __bf16* __restrict__ BT,
                          float* __restrict__ C, int M, int N, int K) {
  const int lane = threadIdx.x;
  const int lo = lane & 15, hi = lane >> 4;
  const int n0 = blockIdx.x * 64;
  const int m0 = blockIdx.y * 32;
  const __bf16* arow0 = A  + (size_t)(m0 + lo) * K + hi * 8;
  const __bf16* arow1 = arow0 + (size_t)16 * K;
  const __bf16* brow  = BT + (size_t)(n0 + lo) * K + hi * 8;

  f32x8 acc[2][4];
#pragma unroll
  for (int i = 0; i < 2; ++i)
#pragma unroll
    for (int j = 0; j < 4; ++j) { f32x8 z = {}; acc[i][j] = z; }

  // prologue: buffer 0
  bf16x16 a0 = load16(arow0);
  bf16x16 a1 = load16(arow1);
  bf16x16 b[4];
#pragma unroll
  for (int j = 0; j < 4; ++j) b[j] = load16(brow + (size_t)j * 16 * K);

#pragma unroll 2
  for (int k = 32; k < K; k += 32) {
    // issue next step's loads first (8 WMMAs of slack before consumption)
    bf16x16 na0 = load16(arow0 + k);
    bf16x16 na1 = load16(arow1 + k);
    bf16x16 nb[4];
#pragma unroll
    for (int j = 0; j < 4; ++j) nb[j] = load16(brow + (size_t)j * 16 * K + k);
    __builtin_prefetch(arow0 + k + 1024, 0, 1);   // -> global_prefetch_b8
    __builtin_prefetch(arow1 + k + 1024, 0, 1);
    __builtin_prefetch(brow + k + 1024, 0, 1);
    __builtin_prefetch(brow + (size_t)48 * K + k + 1024, 0, 1);
    // consume current buffer
#pragma unroll
    for (int j = 0; j < 4; ++j) {
      acc[0][j] = wmma_bf16(a0, b[j], acc[0][j]);
      acc[1][j] = wmma_bf16(a1, b[j], acc[1][j]);
    }
    a0 = na0; a1 = na1;
#pragma unroll
    for (int j = 0; j < 4; ++j) b[j] = nb[j];
  }
  // epilogue: last buffered step
#pragma unroll
  for (int j = 0; j < 4; ++j) {
    acc[0][j] = wmma_bf16(a0, b[j], acc[0][j]);
    acc[1][j] = wmma_bf16(a1, b[j], acc[1][j]);
  }

#pragma unroll
  for (int i = 0; i < 2; ++i)
#pragma unroll
    for (int j = 0; j < 4; ++j) {
      float* crow = C + (size_t)(m0 + i * 16 + hi * 8) * N + n0 + j * 16 + lo;
#pragma unroll
      for (int r = 0; r < 8; ++r) crow[(size_t)r * N] = acc[i][j][r];
    }
}

// ---------------------------------------------------------------------------
// RMS-norm + RoPE for Q: qg (T, H, 2D) fp32 -> Qbf (H, T, D) bf16
__global__ void normrope_q(const float* __restrict__ qg, const float* __restrict__ w,
                           const int* __restrict__ pos, __bf16* __restrict__ Qout) {
  int t = blockIdx.x, h = blockIdx.y, d = threadIdx.x;   // d in [0,64)
  const float* src = qg + (size_t)t * QGN_ + (size_t)h * (2 * D_);
  float x1 = src[d], x2 = src[d + HALF_];
  float ss = x1 * x1 + x2 * x2;
#pragma unroll
  for (int m = 1; m < 32; m <<= 1) ss += __shfl_xor(ss, m, 32);
  __shared__ float part[2];
  if ((threadIdx.x & 31) == 0) part[threadIdx.x >> 5] = ss;
  __syncthreads();
  float inv = rsqrtf((part[0] + part[1]) / (float)D_ + 1e-6f);
  float xn1 = x1 * inv * (1.f + w[d]);
  float xn2 = x2 * inv * (1.f + w[d + HALF_]);
  float p = (float)pos[t];
  float freq = p * __powf(10000.f, -(float)d / (float)HALF_);
  float s, c;
  sincosf(freq, &s, &c);
  __bf16* dst = Qout + ((size_t)h * T_ + t) * D_;
  dst[d]         = tobf(xn1 * c - xn2 * s);
  dst[d + HALF_] = tobf(xn2 * c + xn1 * s);
}

// RMS-norm + RoPE for K: kin (T, KV, D) fp32 -> Kbf (KV, T, D) bf16
__global__ void normrope_k(const float* __restrict__ kin, const float* __restrict__ w,
                           const int* __restrict__ pos, __bf16* __restrict__ Kout) {
  int t = blockIdx.x, kv = blockIdx.y, d = threadIdx.x;  // d in [0,64)
  const float* src = kin + (size_t)t * KVN_ + (size_t)kv * D_;
  float x1 = src[d], x2 = src[d + HALF_];
  float ss = x1 * x1 + x2 * x2;
#pragma unroll
  for (int m = 1; m < 32; m <<= 1) ss += __shfl_xor(ss, m, 32);
  __shared__ float part[2];
  if ((threadIdx.x & 31) == 0) part[threadIdx.x >> 5] = ss;
  __syncthreads();
  float inv = rsqrtf((part[0] + part[1]) / (float)D_ + 1e-6f);
  float xn1 = x1 * inv * (1.f + w[d]);
  float xn2 = x2 * inv * (1.f + w[d + HALF_]);
  float p = (float)pos[t];
  float freq = p * __powf(10000.f, -(float)d / (float)HALF_);
  float s, c;
  sincosf(freq, &s, &c);
  __bf16* dst = Kout + ((size_t)kv * T_ + t) * D_;
  dst[d]         = tobf(xn1 * c - xn2 * s);
  dst[d + HALF_] = tobf(xn2 * c + xn1 * s);
}

// V (T, KV, D) fp32 -> Vt (KV, D, T) bf16  (so PV B-operand is K-contiguous)
__global__ void transpose_v(const float* __restrict__ vbuf, __bf16* __restrict__ Vt) {
  int t = blockIdx.x, kv = blockIdx.y, d = threadIdx.x;  // d in [0,128)
  Vt[((size_t)kv * D_ + d) * T_ + t] =
      tobf(vbuf[(size_t)t * KVN_ + (size_t)kv * D_ + d]);
}

// ---------------------------------------------------------------------------
// Flash attention: one wave per (16-query tile, head). Causal, GQA (rep=2),
// online softmax, sigmoid-gate fused in epilogue. Output bf16 (T, H*D).
__global__ void attn_kernel(const __bf16* __restrict__ Q, const __bf16* __restrict__ Kb,
                            const __bf16* __restrict__ Vt, const float* __restrict__ qg,
                            __bf16* __restrict__ Obf) {
  const int lane = threadIdx.x;
  const int lo = lane & 15, hi = lane >> 4;
  const int q0 = blockIdx.x * 16;
  const int h  = blockIdx.y;
  const int kv = h >> 1;                       // H/KV == 2
  const float scale = 0.08838834764831845f;    // 1/sqrt(128)

  __shared__ __attribute__((aligned(32))) __bf16 pbuf[16 * 32];

  float mrow[8], lrow[8];
  f32x8 Oacc[8];
#pragma unroll
  for (int r = 0; r < 8; ++r) { mrow[r] = -3.0e38f; lrow[r] = 0.f; }
#pragma unroll
  for (int j = 0; j < 8; ++j) { f32x8 z = {}; Oacc[j] = z; }

  // Hoist the 16x128 Q tile (4 A-operands) out of the key loop.
  const __bf16* qrow = Q + ((size_t)h * T_ + q0 + lo) * D_ + hi * 8;
  bf16x16 qa[4];
#pragma unroll
  for (int kk = 0; kk < 4; ++kk) qa[kk] = load16(qrow + kk * 32);

  for (int k0 = 0; k0 <= q0 + 15; k0 += 32) {
    // ---- S = Q Kt for 16x32 key block (two 16x16 C tiles) ----
    f32x8 s0 = {}, s1 = {};
    const __bf16* krow0 = Kb + ((size_t)kv * T_ + k0 + lo) * D_ + hi * 8;
    const __bf16* krow1 = krow0 + (size_t)16 * D_;
    __builtin_prefetch(krow0 + (size_t)32 * D_, 0, 1);   // next key block
    __builtin_prefetch(krow1 + (size_t)32 * D_, 0, 1);
#pragma unroll
    for (int kk = 0; kk < 4; ++kk) {
      s0 = wmma_bf16(qa[kk], load16(krow0 + kk * 32), s0);
      s1 = wmma_bf16(qa[kk], load16(krow1 + kk * 32), s1);
    }
    // ---- scale + causal mask + online softmax ----
    float x0[8], x1[8], tmax[8];
#pragma unroll
    for (int r = 0; r < 8; ++r) {
      int qpos = q0 + r + hi * 8;
      x0[r] = (k0 + lo      <= qpos) ? s0[r] * scale : -3.0e38f;
      x1[r] = (k0 + 16 + lo <= qpos) ? s1[r] * scale : -3.0e38f;
      tmax[r] = fmaxf(x0[r], x1[r]);
    }
#pragma unroll
    for (int m = 1; m < 16; m <<= 1)
#pragma unroll
      for (int r = 0; r < 8; ++r) tmax[r] = fmaxf(tmax[r], __shfl_xor(tmax[r], m, 32));

    float corr[8], rsum[8];
#pragma unroll
    for (int r = 0; r < 8; ++r) {
      float mnew = fmaxf(mrow[r], tmax[r]);
      corr[r] = (mrow[r] > -1.0e38f) ? __expf(mrow[r] - mnew) : 0.f;
      float p0 = (x0[r] > -1.0e38f) ? __expf(x0[r] - mnew) : 0.f;
      float p1 = (x1[r] > -1.0e38f) ? __expf(x1[r] - mnew) : 0.f;
      x0[r] = p0; x1[r] = p1;
      rsum[r] = p0 + p1;
      mrow[r] = mnew;
    }
#pragma unroll
    for (int m = 1; m < 16; m <<= 1)
#pragma unroll
      for (int r = 0; r < 8; ++r) rsum[r] += __shfl_xor(rsum[r], m, 32);
#pragma unroll
    for (int r = 0; r < 8; ++r) lrow[r] = lrow[r] * corr[r] + rsum[r];
#pragma unroll
    for (int j = 0; j < 8; ++j)
#pragma unroll
      for (int r = 0; r < 8; ++r) Oacc[j][r] *= corr[r];

    // ---- P: C-layout -> LDS row-major -> A-layout ----
    __syncthreads();
#pragma unroll
    for (int r = 0; r < 8; ++r) {
      pbuf[(r + hi * 8) * 32 + lo]      = tobf(x0[r]);
      pbuf[(r + hi * 8) * 32 + 16 + lo] = tobf(x1[r]);
    }
    __syncthreads();
    bf16x16 a = load16(pbuf + lo * 32 + hi * 8);

    // ---- O += P V over 8 d-chunks ----
#pragma unroll
    for (int j = 0; j < 8; ++j) {
      const __bf16* vrow = Vt + ((size_t)(kv * D_ + j * 16 + lo)) * T_ + k0 + hi * 8;
      __builtin_prefetch(vrow + 32, 0, 1);               // next key block
      Oacc[j] = wmma_bf16(a, load16(vrow), Oacc[j]);
    }
  }

  // ---- epilogue: 1/l, sigmoid gate, store bf16 ----
#pragma unroll
  for (int j = 0; j < 8; ++j) {
#pragma unroll
    for (int r = 0; r < 8; ++r) {
      int t = q0 + r + hi * 8;
      int d = j * 16 + lo;
      float o = Oacc[j][r] / lrow[r];
      float g = qg[(size_t)t * QGN_ + (size_t)h * (2 * D_) + D_ + d];
      o *= 1.f / (1.f + __expf(-g));
      Obf[(size_t)t * OD_ + h * D_ + d] = tobf(o);
    }
  }
}

// ---------------------------------------------------------------------------
extern "C" void kernel_launch(void* const* d_in, const int* in_sizes, int n_in,
                              void* d_out, int out_size, void* d_ws, size_t ws_size,
                              hipStream_t stream) {
  (void)in_sizes; (void)n_in; (void)out_size; (void)ws_size;
  const int*   positions = (const int*)  d_in[0];
  const float* hs = (const float*)d_in[1];
  const float* Wq = (const float*)d_in[2];
  const float* Wk = (const float*)d_in[3];
  const float* Wv = (const float*)d_in[4];
  const float* Wo = (const float*)d_in[5];
  const float* qw = (const float*)d_in[6];
  const float* kw = (const float*)d_in[7];
  float* out = (float*)d_out;

  char* ws = (char*)d_ws;
  auto carve = [&](size_t bytes) {
    char* p = ws;
    ws += (bytes + 255) & ~(size_t)255;
    return p;
  };
  __bf16* hsb   = (__bf16*)carve((size_t)T_ * HID_ * 2);
  __bf16* WqT   = (__bf16*)carve((size_t)QGN_ * HID_ * 2);
  __bf16* WkT   = (__bf16*)carve((size_t)KVN_ * HID_ * 2);
  __bf16* WvT   = (__bf16*)carve((size_t)KVN_ * HID_ * 2);
  __bf16* WoT   = (__bf16*)carve((size_t)HID_ * OD_ * 2);
  float*  qgbuf = (float*) carve((size_t)T_ * QGN_ * 4);
  float*  kbuf  = (float*) carve((size_t)T_ * KVN_ * 4);
  float*  vbuf  = (float*) carve((size_t)T_ * KVN_ * 4);
  __bf16* Qbf   = (__bf16*)carve((size_t)H_ * T_ * D_ * 2);
  __bf16* Kbf   = (__bf16*)carve((size_t)KV_ * T_ * D_ * 2);
  __bf16* Vtb   = (__bf16*)carve((size_t)KV_ * D_ * T_ * 2);
  __bf16* Obf   = (__bf16*)carve((size_t)T_ * OD_ * 2);

  // 1) precision/layout conversions
  {
    size_t n = (size_t)T_ * HID_;
    cvt_bf16<<<dim3((unsigned)((n + 255) / 256)), dim3(256), 0, stream>>>(hs, hsb, n);
  }
  {
    size_t n = (size_t)HID_ * QGN_;
    transpose_cvt<<<dim3((unsigned)((n + 255) / 256)), dim3(256), 0, stream>>>(Wq, WqT, HID_, QGN_);
  }
  {
    size_t n = (size_t)HID_ * KVN_;
    transpose_cvt<<<dim3((unsigned)((n + 255) / 256)), dim3(256), 0, stream>>>(Wk, WkT, HID_, KVN_);
    transpose_cvt<<<dim3((unsigned)((n + 255) / 256)), dim3(256), 0, stream>>>(Wv, WvT, HID_, KVN_);
  }
  {
    size_t n = (size_t)OD_ * HID_;
    transpose_cvt<<<dim3((unsigned)((n + 255) / 256)), dim3(256), 0, stream>>>(Wo, WoT, OD_, HID_);
  }

  // 2) QKV projections (WMMA), 32x64 tile per wave
  gemm_bf16<<<dim3(QGN_ / 64, T_ / 32), dim3(32), 0, stream>>>(hsb, WqT, qgbuf, T_, QGN_, HID_);
  gemm_bf16<<<dim3(KVN_ / 64, T_ / 32), dim3(32), 0, stream>>>(hsb, WkT, kbuf, T_, KVN_, HID_);
  gemm_bf16<<<dim3(KVN_ / 64, T_ / 32), dim3(32), 0, stream>>>(hsb, WvT, vbuf, T_, KVN_, HID_);

  // 3) RMS-norm + RoPE + layout
  normrope_q<<<dim3(T_, H_),  dim3(HALF_), 0, stream>>>(qgbuf, qw, positions, Qbf);
  normrope_k<<<dim3(T_, KV_), dim3(HALF_), 0, stream>>>(kbuf, kw, positions, Kbf);
  transpose_v<<<dim3(T_, KV_), dim3(D_), 0, stream>>>(vbuf, Vtb);

  // 4) flash attention (WMMA) + gate
  attn_kernel<<<dim3(T_ / 16, H_), dim3(32), 0, stream>>>(Qbf, Kbf, Vtb, qgbuf, Obf);

  // 5) output projection (WMMA) -> fp32 d_out
  gemm_bf16<<<dim3(HID_ / 64, T_ / 32), dim3(32), 0, stream>>>(Obf, WoT, out, T_, HID_, OD_);
}
